// SwitchSparseMLP_86775519248814
// MI455X (gfx1250) — matrix operand order repaired and checked by reading, not compiled, not run on hardware
//
#include <hip/hip_runtime.h>
#include <cstdint>
#include <cstddef>

// ---------------- problem constants ----------------
constexpr int cB = 4, cS = 2048, cD = 1024, cDFF = 4096, cE = 8, cCAP = 320;
#define EPS_F 1e-6f

typedef __attribute__((ext_vector_type(16))) __bf16 v16bf;
typedef __attribute__((ext_vector_type(8)))  __bf16 v8bf;
typedef __attribute__((ext_vector_type(8)))  float  v8f;

static __device__ __forceinline__ v16bf frag16(const __bf16* p0, const __bf16* p1)
{
    v8bf a = *(const v8bf*)p0;     // 16B-aligned -> ds_load_b128
    v8bf b = *(const v8bf*)p1;     // 16B-aligned -> ds_load_b128
    return __builtin_shufflevector(a, b, 0, 1, 2, 3, 4, 5, 6, 7,
                                         8, 9, 10, 11, 12, 13, 14, 15);
}

// ============================================================
// Kernel 0: weight transpose + convert  W[E][K][N] f32 -> BT[E][N][K] bf16
// 32x32 tiles through LDS; all global accesses coalesced
// ============================================================
__global__ __launch_bounds__(256)
void transpose_cvt_kernel(const float* __restrict__ W,
                          __bf16* __restrict__ BT,
                          int K, int N)
{
    const int kt = blockIdx.x;     // K tile
    const int nt = blockIdx.y;     // N tile
    const int e  = blockIdx.z;
    const int tid = threadIdx.x;

    __shared__ __bf16 sT[32][36];  // [k][n], padded row (72B) keeps 8B alignment

    const float* We = W + (size_t)e * K * N;
    __bf16* Be      = BT + (size_t)e * N * K;

    // load 32x32 f32 tile, convert, store 4 bf16 (8B) per thread
    {
        const int r  = tid >> 3;          // k offset 0..31
        const int c0 = (tid & 7) * 4;     // n offset
        const float4 v = *(const float4*)(We + (size_t)(kt * 32 + r) * N + nt * 32 + c0);
        __bf16 t4[4] = { (__bf16)v.x, (__bf16)v.y, (__bf16)v.z, (__bf16)v.w };
        *(uint2*)&sT[r][c0] = *(const uint2*)t4;
    }
    __syncthreads();
    // write transposed: 4 consecutive k per thread (8B store), coalesced in k
    {
        const int n  = tid >> 3;          // n offset 0..31
        const int k0 = (tid & 7) * 4;     // k offset
        __bf16 t4[4] = { sT[k0][n], sT[k0 + 1][n], sT[k0 + 2][n], sT[k0 + 3][n] };
        *(uint2*)(Be + (size_t)(nt * 32 + n) * K + kt * 32 + k0) = *(const uint2*)t4;
    }
}

// ============================================================
// Kernel 1: RMSNorm (T5 style) + router logits + softmax top-1
// ============================================================
__global__ __launch_bounds__(256)
void rms_router_kernel(const float* __restrict__ hs,
                       const float* __restrict__ lnw,
                       const float* __restrict__ rw,      // [D][E]
                       float* __restrict__ fwd,
                       int*   __restrict__ eidx,
                       float* __restrict__ rp)
{
    const int token = blockIdx.x;            // 0 .. B*S-1
    const int tid   = threadIdx.x;           // 0 .. 255
    const size_t base = (size_t)token * cD;

    __shared__ float sred[256];
    __shared__ float slog[cE];

    const float4 xv = *(const float4*)(hs + base + tid * 4);
    float x[4] = { xv.x, xv.y, xv.z, xv.w };
    float ss = x[0]*x[0] + x[1]*x[1] + x[2]*x[2] + x[3]*x[3];

    sred[tid] = ss;
    __syncthreads();
    for (int off = 128; off > 0; off >>= 1) {
        if (tid < off) sred[tid] += sred[tid + off];
        __syncthreads();
    }
    const float inv = rsqrtf(sred[0] / (float)cD + EPS_F);

    if (tid < cE) slog[tid] = 0.f;
    __syncthreads();

    float pl[cE];
#pragma unroll
    for (int e = 0; e < cE; ++e) pl[e] = 0.f;

    float f4[4];
#pragma unroll
    for (int j = 0; j < 4; ++j) {
        const int i = tid * 4 + j;
        const float f = x[j] * inv * lnw[i];
        f4[j] = f;
#pragma unroll
        for (int e = 0; e < cE; ++e) pl[e] += f * rw[(size_t)i * cE + e];
    }
    *(float4*)(fwd + base + tid * 4) = make_float4(f4[0], f4[1], f4[2], f4[3]);

#pragma unroll
    for (int e = 0; e < cE; ++e) {
        float v = pl[e];
        for (int off = 16; off > 0; off >>= 1) v += __shfl_down(v, off, 32);
        if ((tid & 31) == 0) atomicAdd(&slog[e], v);
    }
    __syncthreads();

    if (tid == 0) {
        float mx = slog[0]; int am = 0;
        for (int e = 1; e < cE; ++e) { if (slog[e] > mx) { mx = slog[e]; am = e; } }
        float sum = 0.f;
        for (int e = 0; e < cE; ++e) sum += __expf(slog[e] - mx);
        eidx[token] = am;
        rp[token]   = 1.0f / sum;     // max softmax prob
    }
}

// ============================================================
// Kernel 2: per-batch capacity scan (cumsum of one-hot), one wave32/batch
// ============================================================
__global__ __launch_bounds__(32)
void route_scan_kernel(const int* __restrict__ eidx,
                       int* __restrict__ pos,
                       int* __restrict__ route,
                       int* __restrict__ cnt)
{
    const int b    = blockIdx.x;
    const int lane = threadIdx.x;
    const unsigned ltmask = (lane == 0) ? 0u : ((1u << lane) - 1u);

    int base[cE];
#pragma unroll
    for (int e = 0; e < cE; ++e) base[e] = 0;

    for (int chunk = 0; chunk < cS / 32; ++chunk) {
        const int s  = chunk * 32 + lane;
        const int my = eidx[b * cS + s];
        int myPos = 0;
#pragma unroll
        for (int e = 0; e < cE; ++e) {
            const unsigned m = (unsigned)__ballot(my == e);
            if (my == e) myPos = base[e] + __popc(m & ltmask);
            base[e] += __popc(m);
        }
        pos[b * cS + s] = myPos;
        if (myPos < cCAP)
            route[((size_t)my * cB + b) * cCAP + myPos] = s;
    }
    if (lane < cE) cnt[lane * cB + b] = base[lane];
}

// ============================================================
// Kernel 3: dispatch gather -> zero-padded bf16 A matrix [E][B*CAP][D]
// ============================================================
__global__ __launch_bounds__(256)
void dispatch_kernel(const float* __restrict__ fwd,
                     const int*   __restrict__ route,
                     const int*   __restrict__ cnt,
                     __bf16*      __restrict__ disp)
{
    const int row = blockIdx.x;               // 0 .. E*B*CAP-1
    const int tid = threadIdx.x;
    const int e    = row / (cB * cCAP);
    const int r2   = row - e * cB * cCAP;
    const int b    = r2 / cCAP;
    const int slot = r2 - b * cCAP;
    __bf16* dst = disp + (size_t)row * cD;

    __bf16 v[4];
    if (slot < cnt[e * cB + b]) {
        const int s = route[((size_t)e * cB + b) * cCAP + slot];
        const float4 sv = *(const float4*)(fwd + ((size_t)b * cS + s) * cD + tid * 4);
        v[0] = (__bf16)sv.x; v[1] = (__bf16)sv.y;
        v[2] = (__bf16)sv.z; v[3] = (__bf16)sv.w;
    } else {
        v[0] = v[1] = v[2] = v[3] = (__bf16)0.0f;
    }
    *(uint2*)(dst + tid * 4) = *(const uint2*)v;   // 8-byte store
}

// ============================================================
// Kernels 4/5: grouped GEMM via v_wmma_f32_16x16x32_bf16
// block = 256 threads (8 waves, 4 row x 2 col); block tile 128x128, BK=32
// each wave: 32x64 strip = 8 wmma accumulators (2 row x 4 col tiles)
// A:  [E][M][K] bf16 row-major
// BT: [E][N][K] bf16 (pre-transposed weights)
// C:  [E][M][N] bf16 (ReLU, stage 1) or f32 (stage 2)
// ============================================================
template<int K, int N, bool RELU, bool OUT_BF16>
__global__ __launch_bounds__(256)
void moe_gemm_kernel(const __bf16* __restrict__ Aall,
                     const __bf16* __restrict__ BTall,
                     void*         __restrict__ Call,
                     int M)
{
    const int rt   = blockIdx.x;       // 128-row tile
    const int ct   = blockIdx.y;       // 128-col tile
    const int e    = blockIdx.z;       // expert
    const int tid  = threadIdx.x;
    const int wave = tid >> 5;
    const int lane = tid & 31;
    const int n16  = lane & 15;        // fragment column / A row
    const int h    = lane >> 4;        // half-wave
    const int wr   = wave >> 1;        // wave row 0..3 (32 rows each)
    const int wc   = wave & 1;         // wave col 0..1 (64 cols each)

    const int rowBase = rt * 128;
    const int colBase = ct * 128;

    const __bf16* Ae = Aall  + (size_t)e * M * K;
    const __bf16* Te = BTall + (size_t)e * N * K;

    __shared__ __bf16 sA[128][32];       // 8 KB, [m][k]
    __shared__ __bf16 sBt[128][40];      // 10 KB, [n][k] padded to 80B rows

    v8f acc[2][4] = {{{}, {}, {}, {}}, {{}, {}, {}, {}}};

    for (int kk = 0; kk < K / 32; ++kk) {
        if (kk + 1 < K / 32) {          // speculative prefetch of next k-slab
            __builtin_prefetch(Ae + (size_t)(rowBase + (tid >> 1)) * K
                                  + (kk + 1) * 32 + (tid & 1) * 16, 0, 1);
            __builtin_prefetch(Te + (size_t)(colBase + (tid >> 1)) * K
                                  + (kk + 1) * 32 + (tid & 1) * 16, 0, 1);
        }

        __syncthreads();
        // ---- stage A: 128x32 bf16 = 512 x 16B chunks, 2 per thread ----
#pragma unroll
        for (int j = 0; j < 2; ++j) {
            const int chunk = tid + 256 * j;
            const int r = chunk >> 2;
            const int q = chunk & 3;
            *(uint4*)&sA[r][q * 8] =
                *(const uint4*)(Ae + (size_t)(rowBase + r) * K + kk * 32 + q * 8);
        }
        // ---- stage BT: 128x32 bf16 = 512 x 16B chunks, 2 per thread ----
#pragma unroll
        for (int j = 0; j < 2; ++j) {
            const int chunk = tid + 256 * j;
            const int c = chunk >> 2;
            const int q = chunk & 3;
            *(uint4*)&sBt[c][q * 8] =
                *(const uint4*)(Te + (size_t)(colBase + c) * K + kk * 32 + q * 8);
        }
        __syncthreads();

        // ---- fragments (all 2x ds_load_b128) + 8 wmma ----
        v16bf af[2];
#pragma unroll
        for (int i = 0; i < 2; ++i) {
            const int arow = wr * 32 + i * 16 + n16;
            af[i] = frag16(&sA[arow][8 * h], &sA[arow][16 + 8 * h]);
        }
#pragma unroll
        for (int t = 0; t < 4; ++t) {
            const int bcol = wc * 64 + t * 16 + n16;
            const v16bf bfrag = frag16(&sBt[bcol][8 * h], &sBt[bcol][16 + 8 * h]);
#pragma unroll
            for (int i = 0; i < 2; ++i) {
                acc[i][t] = __builtin_amdgcn_wmma_f32_16x16x32_bf16(
                                false, af[i], false, bfrag, (short)0,
                                acc[i][t], false, false);
            }
        }
    }

    // ---- store C: lane n = lane&15, VGPR v -> row v + 8*h ----
#pragma unroll
    for (int i = 0; i < 2; ++i) {
#pragma unroll
        for (int t = 0; t < 4; ++t) {
            const int n = colBase + wc * 64 + t * 16 + n16;
#pragma unroll
            for (int v = 0; v < 8; ++v) {
                const int row = rowBase + wr * 32 + i * 16 + v + 8 * h;
                float val = acc[i][t][v];
                if (RELU) val = fmaxf(val, 0.0f);
                if (OUT_BF16)
                    ((__bf16*)Call)[(size_t)e * M * N + (size_t)row * N + n] = (__bf16)val;
                else
                    ((float*)Call)[(size_t)e * M * N + (size_t)row * N + n] = val;
            }
        }
    }
}

// ============================================================
// Kernel 6: combine  out = hs + rp * (keep ? eout[slot] : fwd)
// ============================================================
__global__ __launch_bounds__(256)
void combine_kernel(const float* __restrict__ hs,
                    const float* __restrict__ fwd,
                    const float* __restrict__ eout,
                    const int*   __restrict__ eidx,
                    const int*   __restrict__ pos,
                    const float* __restrict__ rp,
                    float* __restrict__ out)
{
    const int token = blockIdx.x;
    const int tid   = threadIdx.x;
    const int b = token / cS;
    const int e = eidx[token];
    const int p = pos[token];
    const bool keep = (p < cCAP);
    const float r = rp[token];
    const size_t base = (size_t)token * cD + tid * 4;
    const size_t erow = (((size_t)e * cB + b) * cCAP + (keep ? p : (cCAP - 1)))
                        * (size_t)cD + tid * 4;

    const float4 hv = *(const float4*)(hs + base);
    const float4 sv = keep ? *(const float4*)(eout + erow)
                           : *(const float4*)(fwd + base);
    float4 o;
    o.x = hv.x + r * sv.x;
    o.y = hv.y + r * sv.y;
    o.z = hv.z + r * sv.z;
    o.w = hv.w + r * sv.w;
    *(float4*)(out + base) = o;
}

// ============================================================
// launch
// ============================================================
extern "C" void kernel_launch(void* const* d_in, const int* in_sizes, int n_in,
                              void* d_out, int out_size, void* d_ws, size_t ws_size,
                              hipStream_t stream)
{
    (void)in_sizes; (void)n_in; (void)out_size; (void)ws_size;

    const float* hs  = (const float*)d_in[0];
    const float* lnw = (const float*)d_in[1];
    const float* rw  = (const float*)d_in[2];
    const float* wi  = (const float*)d_in[3];
    const float* wo  = (const float*)d_in[4];
    float* out = (float*)d_out;

    char* ws = (char*)d_ws;
    size_t off = 0;
    auto alloc = [&](size_t bytes) -> char* {
        char* p = ws + off;
        off += (bytes + 255) & ~(size_t)255;
        return p;
    };

    float*  fwd   = (float*) alloc((size_t)cB * cS * cD * 4);            // 32 MB
    int*    eidx  = (int*)   alloc((size_t)cB * cS * 4);
    float*  rp    = (float*) alloc((size_t)cB * cS * 4);
    int*    pos   = (int*)   alloc((size_t)cB * cS * 4);
    int*    route = (int*)   alloc((size_t)cE * cB * cCAP * 4);
    int*    cnt   = (int*)   alloc((size_t)cE * cB * 4);
    __bf16* disp  = (__bf16*)alloc((size_t)cE * cB * cCAP * cD * 2);     // 21 MB
    __bf16* H     = (__bf16*)alloc((size_t)cE * cB * cCAP * cDFF * 2);   // 84 MB
    float*  eout  = (float*) alloc((size_t)cE * cB * cCAP * cD * 4);     // 42 MB
    __bf16* wiT   = (__bf16*)alloc((size_t)cE * cD * cDFF * 2);          // 67 MB [E][DFF][D]
    __bf16* woT   = (__bf16*)alloc((size_t)cE * cDFF * cD * 2);          // 67 MB [E][D][DFF]

    // 0) one-shot weight transpose+convert: W[E][K][N] f32 -> BT[E][N][K] bf16
    transpose_cvt_kernel<<<dim3(cD / 32, cDFF / 32, cE), 256, 0, stream>>>(
        wi, wiT, cD, cDFF);
    transpose_cvt_kernel<<<dim3(cDFF / 32, cD / 32, cE), 256, 0, stream>>>(
        wo, woT, cDFF, cD);

    // 1) RMSNorm + router
    rms_router_kernel<<<cB * cS, 256, 0, stream>>>(hs, lnw, rw, fwd, eidx, rp);

    // 2) capacity scan (exact cumsum semantics)
    route_scan_kernel<<<cB, 32, 0, stream>>>(eidx, pos, route, cnt);

    // 3) dispatch into zero-padded bf16 A matrices
    dispatch_kernel<<<cE * cB * cCAP, 256, 0, stream>>>(fwd, route, cnt, disp);

    // 4) H = relu(disp @ wi)      M=1280, K=1024, N=4096
    const int M = cB * cCAP;                    // 1280, divisible by 128? no: 1280/128=10 exact
    moe_gemm_kernel<cD, cDFF, true, true>
        <<<dim3(M / 128, cDFF / 128, cE), 256, 0, stream>>>(disp, wiT, (void*)H, M);

    // 5) eout = H @ wo            M=1280, K=4096, N=1024
    moe_gemm_kernel<cDFF, cD, false, false>
        <<<dim3(M / 128, cD / 128, cE), 256, 0, stream>>>(H, woT, (void*)eout, M);

    // 6) combine
    combine_kernel<<<cB * cS, 256, 0, stream>>>(hs, fwd, eout, eidx, pos, rp, out);
}
